// BinConvBnHTanh_9491877724614
// MI455X (gfx1250) — 8-bit matrix operand order repaired and checked by reading, compiled, not run on hardware
//
#include <hip/hip_runtime.h>
#include <stdint.h>

typedef int v8i __attribute__((ext_vector_type(8)));

#define N_IMG   32
#define C_IN    256
#define H_IMG   56
#define W_IMG   56
#define C_OUT   256
#define HW      (H_IMG*W_IMG)
#define K_TOT   (C_IN*9)
#define K_STEPS (K_TOT/64)
#define BM      128
#define BN      128
#define M_TOT   (N_IMG*HW)
#define STG_W   20

#if __has_builtin(__builtin_amdgcn_global_load_async_to_lds_b128) && __has_builtin(__builtin_amdgcn_s_wait_asynccnt)
#define USE_ASYNC_LDS 1
typedef int v4i_vs __attribute__((vector_size(16)));
typedef __attribute__((address_space(1))) v4i_vs GV4;
typedef __attribute__((address_space(3))) v4i_vs LV4;
#endif

__global__ __launch_bounds__(256) void binx_kernel(const float* __restrict__ x,
                                                   int8_t* __restrict__ xb) {
    int idx = blockIdx.x * 256 + threadIdx.x;
    int c   = idx & (C_IN - 1);
    int p   = idx >> 8;
    int n   = p / HW;
    int hw  = p - n * HW;
    float v = x[((n * C_IN + c) * HW) + hw];
    xb[idx] = (int8_t)((v > 0.0f) - (v < 0.0f));
}

__global__ __launch_bounds__(256) void binw_kernel(const float* __restrict__ w,
                                                   int8_t* __restrict__ wb) {
    int idx = blockIdx.x * 256 + threadIdx.x;
    int k   = idx % K_TOT;
    int co  = idx / K_TOT;
    int tap = k >> 8;
    int ci  = k & (C_IN - 1);
    float v = w[(co * C_IN + ci) * 9 + tap];
    wb[idx] = (int8_t)((v > 0.0f) - (v < 0.0f));
}

__global__ __launch_bounds__(256) void bingemm_kernel(
        const int8_t* __restrict__ xb, const int8_t* __restrict__ wb,
        const float* __restrict__ gamma, const float* __restrict__ beta,
        const float* __restrict__ rmean, const float* __restrict__ rvar,
        float* __restrict__ out) {

    __shared__ int8_t As[2][BM * 64];
    __shared__ int8_t Bs[2][BN * 64];
    __shared__ float  sscale[BN];
    __shared__ float  sbias[BN];

    const int tid    = threadIdx.x;
    const int lane   = tid & 31;
    const int wid    = tid >> 5;
    const int wave_m = wid >> 1;
    const int wave_n = wid & 1;
    const int half   = lane >> 4;
    const int l16    = lane & 15;

    const int bn = blockIdx.x & 1;
    const int bm = blockIdx.x >> 1;

    if (tid < BN) {
        const int c = bn * BN + tid;
        const float sc = gamma[c] * rsqrtf(rvar[c] + 1e-5f);
        sscale[tid] = sc;
        sbias[tid]  = beta[c] - rmean[c] * sc;
    }

    const int lrow = tid >> 1;
    const int lseg = (tid & 1) * 32;

    const int pA   = bm * BM + lrow;
    const int nImg = pA / HW;
    const int hwA  = pA - nImg * HW;
    const int hA   = hwA / W_IMG;
    const int wA   = hwA - hA * W_IMG;
    const int cB   = bn * BN + lrow;

    v8i acc[2][4];
    #pragma unroll
    for (int tm = 0; tm < 2; ++tm)
        #pragma unroll
        for (int tn = 0; tn < 4; ++tn)
            acc[tm][tn] = (v8i){0, 0, 0, 0, 0, 0, 0, 0};

    auto load_tiles = [&](int ks, int buf) {
        const int tap = ks >> 2;
        const int cb  = ks & 3;
        const int hs  = hA + (tap / 3) - 1;
        const int ws  = wA + (tap % 3) - 1;
        int8_t* adst = &As[buf][lrow * 64 + lseg];
        int8_t* bdst = &Bs[buf][lrow * 64 + lseg];
        const int8_t* bsrc = wb + cB * K_TOT + ks * 64 + lseg;
        const bool inb = (hs >= 0) & (hs < H_IMG) & (ws >= 0) & (ws < W_IMG);
#ifdef USE_ASYNC_LDS
        __builtin_amdgcn_global_load_async_to_lds_b128((GV4*)bsrc,        (LV4*)bdst,        0, 0);
        __builtin_amdgcn_global_load_async_to_lds_b128((GV4*)(bsrc + 16), (LV4*)(bdst + 16), 0, 0);
        if (inb) {
            const int8_t* asrc = xb + ((nImg * H_IMG + hs) * W_IMG + ws) * C_IN + cb * 64 + lseg;
            __builtin_amdgcn_global_load_async_to_lds_b128((GV4*)asrc,        (LV4*)adst,        0, 0);
            __builtin_amdgcn_global_load_async_to_lds_b128((GV4*)(asrc + 16), (LV4*)(adst + 16), 0, 0);
        } else {
            const int4 z = make_int4(0, 0, 0, 0);
            *(int4*)(adst)      = z;
            *(int4*)(adst + 16) = z;
        }
#else
        int4 b0 = *(const int4*)bsrc;
        int4 b1 = *(const int4*)(bsrc + 16);
        int4 a0 = make_int4(0, 0, 0, 0), a1 = a0;
        if (inb) {
            const int8_t* asrc = xb + ((nImg * H_IMG + hs) * W_IMG + ws) * C_IN + cb * 64 + lseg;
            a0 = *(const int4*)asrc;
            a1 = *(const int4*)(asrc + 16);
        }
        *(int4*)(bdst)      = b0;
        *(int4*)(bdst + 16) = b1;
        *(int4*)(adst)      = a0;
        *(int4*)(adst + 16) = a1;
#endif
    };

    auto mma_step = [&](int buf) {
        v8i a[2], b[4];
        #pragma unroll
        for (int tm = 0; tm < 2; ++tm) {
            const int8_t* ar = &As[buf][(wave_m * 32 + tm * 16 + l16) * 64 + half * 16];
            int4 p0 = *(const int4*)(ar);
            int4 p1 = *(const int4*)(ar + 32);
            a[tm] = (v8i){p0.x, p0.y, p0.z, p0.w, p1.x, p1.y, p1.z, p1.w};
        }
        #pragma unroll
        for (int tn = 0; tn < 4; ++tn) {
            const int8_t* br = &Bs[buf][(wave_n * 64 + tn * 16 + l16) * 64 + half * 16];
            int4 q0 = *(const int4*)(br);
            int4 q1 = *(const int4*)(br + 32);
            b[tn] = (v8i){q0.x, q0.y, q0.z, q0.w, q1.x, q1.y, q1.z, q1.w};
        }
        #pragma unroll
        for (int tm = 0; tm < 2; ++tm)
            #pragma unroll
            for (int tn = 0; tn < 4; ++tn)
                acc[tm][tn] = __builtin_amdgcn_wmma_i32_16x16x64_iu8(
                    true, a[tm], true, b[tn], acc[tm][tn], false, false);
    };

    load_tiles(0, 0);
#ifdef USE_ASYNC_LDS
    __builtin_amdgcn_s_wait_asynccnt(0);
#endif
    __syncthreads();
    for (int ks = 0; ks < K_STEPS; ++ks) {
        if (ks + 1 < K_STEPS) load_tiles(ks + 1, (ks + 1) & 1);
        mma_step(ks & 1);
#ifdef USE_ASYNC_LDS
        __builtin_amdgcn_s_wait_asynccnt(0);
#endif
        __syncthreads();
    }

    float* stage = (float*)&As[0][0] + wid * (16 * STG_W);
    #pragma unroll
    for (int tm = 0; tm < 2; ++tm) {
        #pragma unroll
        for (int tn = 0; tn < 4; ++tn) {
            #pragma unroll
            for (int r = 0; r < 8; ++r)
                stage[(half * 8 + r) * STG_W + l16] = (float)acc[tm][tn][r];
            #pragma unroll
            for (int i = 0; i < 8; ++i) {
                const int c_in = half + 2 * i;
                const float v  = stage[l16 * STG_W + c_in];
                const int cl   = wave_n * 64 + tn * 16 + c_in;
                const int m    = bm * BM + wave_m * 32 + tm * 16 + l16;
                const int n    = m / HW;
                const int hw   = m - n * HW;
                out[(n * C_OUT + bn * BN + cl) * HW + hw] = v * sscale[cl] + sbias[cl];
            }
        }
    }
}

extern "C" void kernel_launch(void* const* d_in, const int* in_sizes, int n_in,
                              void* d_out, int out_size, void* d_ws, size_t ws_size,
                              hipStream_t stream) {
    const float* x     = (const float*)d_in[0];
    const float* w     = (const float*)d_in[1];
    const float* gamma = (const float*)d_in[2];
    const float* beta  = (const float*)d_in[3];
    const float* rmean = (const float*)d_in[4];
    const float* rvar  = (const float*)d_in[5];
    float* out = (float*)d_out;

    int8_t* xb = (int8_t*)d_ws;
    int8_t* wb = xb + (size_t)N_IMG * HW * C_IN;

    binx_kernel<<<M_TOT * C_IN / 256, 256, 0, stream>>>(x, xb);
    binw_kernel<<<C_OUT * K_TOT / 256, 256, 0, stream>>>(w, wb);
    bingemm_kernel<<<(M_TOT / BM) * (C_OUT / BN), 256, 0, stream>>>(
        xb, wb, gamma, beta, rmean, rvar, out);
}
